// SelfAttentionWithGate_70239895159355
// MI455X (gfx1250) — matrix operand order repaired and checked by reading, not compiled, not run on hardware
//
#include <hip/hip_runtime.h>

// ---- problem dims ----
#define NUM_HEADS 8
#define C_HIDDEN  32
#define RDIM      64
#define SDIM      512
#define NROWS     (RDIM * SDIM)          // 32768 tokens
#define DMODEL    (NUM_HEADS * C_HIDDEN) // 256
#define CQKV      256

typedef __bf16 bf16_t;
typedef __bf16 bf16x8  __attribute__((ext_vector_type(8)));
typedef __bf16 bf16x16 __attribute__((ext_vector_type(16)));
typedef float  f32x8   __attribute__((ext_vector_type(8)));

// A-fragment (16x32 bf16, MxK): lane holds row (lane&15); K = half*16+[0..15]
// contiguous in elements 0..15 -> two 16B loads.
__device__ __forceinline__ bf16x16 load_afrag(const bf16_t* p) {
  union { bf16x16 v; bf16x8 h[2]; } u;
  u.h[0] = *(const bf16x8*)(p);
  u.h[1] = *(const bf16x8*)(p + 8);
  return u.v;
}
// B-fragment (32x16 bf16, KxN): lane holds col (lane&15); K = half*16+[0..15]
// contiguous in elements 0..15 -> two 16B loads.
__device__ __forceinline__ bf16x16 load_bfrag(const bf16_t* p) {
  union { bf16x16 v; bf16x8 h[2]; } u;
  u.h[0] = *(const bf16x8*)(p);
  u.h[1] = *(const bf16x8*)(p + 8);
  return u.v;
}

__device__ __forceinline__ f32x8 wmma_bf16(bf16x16 a, bf16x16 b, f32x8 c) {
  return __builtin_amdgcn_wmma_f32_16x16x32_bf16(false, a, false, b,
                                                 (short)0, c, false, false);
}

// reductions across the 16 lanes of one half-wave (row of a C-fragment)
__device__ __forceinline__ float hmax16(float x) {
  x = fmaxf(x, __shfl_xor(x, 1, 32));
  x = fmaxf(x, __shfl_xor(x, 2, 32));
  x = fmaxf(x, __shfl_xor(x, 4, 32));
  x = fmaxf(x, __shfl_xor(x, 8, 32));
  return x;
}
__device__ __forceinline__ float hadd16(float x) {
  x += __shfl_xor(x, 1, 32);
  x += __shfl_xor(x, 2, 32);
  x += __shfl_xor(x, 4, 32);
  x += __shfl_xor(x, 8, 32);
  return x;
}

// ---------------- kernel 1: fp32 -> bf16 convert ----------------
__global__ void cvt_kernel(const float* __restrict__ src, bf16_t* __restrict__ dst, int n) {
  int i = blockIdx.x * blockDim.x + threadIdx.x;
  if (i < n) dst[i] = (bf16_t)src[i];
}

// ---------------- kernel 2: weight transpose + convert (Wt[n][k]) ----------------
__global__ void wtrans_kernel(const float* __restrict__ w, bf16_t* __restrict__ wt, float scale) {
  int idx = blockIdx.x * 256 + threadIdx.x;   // idx = k*256 + n
  int k = idx >> 8, n = idx & 255;
  wt[n * 256 + k] = (bf16_t)(w[idx] * scale);
}

// ---------------- kernel 3: projections (Q scaled, K, V^T, gate) ----------------
// One wave computes a 64x64 register tile: 4 A-frags x 4 B-frags -> 16 WMMA per
// k-step with only 8 fragment loads (2:1 wmma:load), ~4x less L2 traffic.
__global__ __launch_bounds__(32) void proj_kernel(
    const bf16_t* __restrict__ xb,
    const bf16_t* __restrict__ wtq, const bf16_t* __restrict__ wtk,
    const bf16_t* __restrict__ wtv, const bf16_t* __restrict__ wtg,
    const float* __restrict__ bg,
    bf16_t* __restrict__ qO, bf16_t* __restrict__ kO,
    bf16_t* __restrict__ vtO, float* __restrict__ gateO) {
  const int lane = threadIdx.x & 31;
  const int half = lane >> 4;
  const int l15  = lane & 15;
  const int mt   = blockIdx.x;       // 0..511  (64-row tile of tokens)
  const int nt   = blockIdx.y;       // 0..3    (64-col tile of d_model)
  const int mode = blockIdx.z;       // 0=Q 1=K 2=V 3=gate

  const bf16_t* wt = (mode == 0) ? wtq : (mode == 1) ? wtk : (mode == 2) ? wtv : wtg;

  f32x8 acc[4][4];
  {
    f32x8 z = {};
#pragma unroll
    for (int i = 0; i < 4; ++i)
#pragma unroll
      for (int j = 0; j < 4; ++j) acc[i][j] = z;
  }

  const bf16_t* ap = xb + (mt * 64 + l15) * 256 + half * 16;  // +i*16*256
  const bf16_t* bp = wt + (nt * 64 + l15) * 256 + half * 16;  // +j*16*256

#pragma unroll
  for (int kk = 0; kk < 256; kk += 32) {
    bf16x16 a[4], b[4];
#pragma unroll
    for (int i = 0; i < 4; ++i) a[i] = load_afrag(ap + i * 16 * 256 + kk);
#pragma unroll
    for (int j = 0; j < 4; ++j) b[j] = load_bfrag(bp + j * 16 * 256 + kk);
#pragma unroll
    for (int i = 0; i < 4; ++i)
#pragma unroll
      for (int j = 0; j < 4; ++j) acc[i][j] = wmma_bf16(a[i], b[j], acc[i][j]);
  }

#pragma unroll
  for (int i = 0; i < 4; ++i) {
#pragma unroll
    for (int j = 0; j < 4; ++j) {
      const int n = nt * 64 + j * 16 + l15;
      const int h = n >> 5, d = n & 31;
#pragma unroll
      for (int v = 0; v < 8; ++v) {
        int m = mt * 64 + i * 16 + v + 8 * half;    // global token row
        int r = m >> 9, s = m & 511;
        float x = acc[i][j][v];
        if (mode == 0)      qO[((r * 8 + h) * 512 + s) * 32 + d] = (bf16_t)x;
        else if (mode == 1) kO[((r * 8 + h) * 512 + s) * 32 + d] = (bf16_t)x;
        else if (mode == 2) vtO[((r * 8 + h) * 32 + d) * 512 + s] = (bf16_t)x;
        else                gateO[m * 256 + n] = 1.0f / (1.0f + __expf(-(x + bg[n])));
      }
    }
  }
}

// ---------------- kernel 4: flash attention + gate ----------------
__global__ __launch_bounds__(128) void attn_kernel(
    const bf16_t* __restrict__ qp, const bf16_t* __restrict__ kp,
    const bf16_t* __restrict__ vtp, const float* __restrict__ bias,
    const float* __restrict__ mask, const float* __restrict__ gate,
    bf16_t* __restrict__ og) {
  __shared__ __align__(16) bf16_t plds[4][16 * 32];  // per-wave P staging
  const int lane = threadIdx.x & 31;
  const int wid  = threadIdx.x >> 5;
  const int half = lane >> 4;
  const int l15  = lane & 15;
  const int h  = blockIdx.y;
  const int r  = blockIdx.z;
  const int q0 = blockIdx.x * 64 + wid * 16;   // this wave's 16 q-rows

  const bf16_t* Q  = qp  + (r * 8 + h) * 512 * 32;  // [s][d]
  const bf16_t* K  = kp  + (r * 8 + h) * 512 * 32;  // [s][d]
  const bf16_t* VT = vtp + (r * 8 + h) * 32 * 512;  // [d][s]
  const float*  Bh = bias + (h * 512 + q0) * 512;   // [q][k]
  const float*  Mr = mask + r * 512;

  bf16x16 aq = load_afrag(Q + (q0 + l15) * 32 + half * 16);  // resident A-fragment

  float mrow[8], lrow[8];
  f32x8 o0 = {}, o1 = {};   // 16x32 output accum (d tiles 0..15 / 16..31)
#pragma unroll
  for (int v = 0; v < 8; ++v) { mrow[v] = -1e30f; lrow[v] = 0.0f; }

  for (int k0 = 0; k0 < 512; k0 += 32) {
    f32x8 z = {};
    bf16x16 bk0 = load_bfrag(K + (k0 + l15) * 32 + half * 16);
    bf16x16 bk1 = load_bfrag(K + (k0 + 16 + l15) * 32 + half * 16);
    f32x8 c0 = wmma_bf16(aq, bk0, z);   // scores q-tile x keys[k0..k0+15]
    f32x8 c1 = wmma_bf16(aq, bk1, z);   // scores q-tile x keys[k0+16..k0+31]

    float mk0 = (Mr[k0 + l15] - 1.0f) * 1e9f;
    float mk1 = (Mr[k0 + 16 + l15] - 1.0f) * 1e9f;
#pragma unroll
    for (int v = 0; v < 8; ++v) {
      int row = v + 8 * half;
      float s0 = c0[v] + Bh[row * 512 + k0 + l15] + mk0;
      float s1 = c1[v] + Bh[row * 512 + k0 + 16 + l15] + mk1;
      float t  = hmax16(fmaxf(s0, s1));
      float mn = fmaxf(mrow[v], t);
      float sc = __expf(mrow[v] - mn);
      float p0 = __expf(s0 - mn);
      float p1 = __expf(s1 - mn);
      lrow[v] = lrow[v] * sc + hadd16(p0 + p1);
      mrow[v] = mn;
      o0[v] *= sc;
      o1[v] *= sc;
      plds[wid][row * 32 + l15]      = (bf16_t)p0;   // P tile [q][k] in LDS
      plds[wid][row * 32 + 16 + l15] = (bf16_t)p1;
    }
    __syncthreads();
    bf16x16 pa = load_afrag(&plds[wid][l15 * 32 + half * 16]);  // P as A-fragment
    __syncthreads();
    bf16x16 bv0 = load_bfrag(VT + l15 * 512 + k0 + half * 16);        // d 0..15
    bf16x16 bv1 = load_bfrag(VT + (16 + l15) * 512 + k0 + half * 16); // d 16..31
    o0 = wmma_bf16(pa, bv0, o0);
    o1 = wmma_bf16(pa, bv1, o1);
  }

#pragma unroll
  for (int v = 0; v < 8; ++v) {
    int row = v + 8 * half;
    int m   = r * 512 + q0 + row;    // global token
    float inv = 1.0f / lrow[v];
    int cA = h * 32 + l15;
    int cB = h * 32 + 16 + l15;
    og[m * 256 + cA] = (bf16_t)(o0[v] * inv * gate[m * 256 + cA]);
    og[m * 256 + cB] = (bf16_t)(o1[v] * inv * gate[m * 256 + cB]);
  }
}

// ---------------- kernel 5: output projection (64x64 register tile) ----------------
__global__ __launch_bounds__(32) void outgemm_kernel(
    const bf16_t* __restrict__ og, const bf16_t* __restrict__ wto,
    const float* __restrict__ bo, float* __restrict__ out) {
  const int lane = threadIdx.x & 31;
  const int half = lane >> 4;
  const int l15  = lane & 15;
  const int mt   = blockIdx.x;   // 0..511
  const int nt   = blockIdx.y;   // 0..3

  f32x8 acc[4][4];
  {
    f32x8 z = {};
#pragma unroll
    for (int i = 0; i < 4; ++i)
#pragma unroll
      for (int j = 0; j < 4; ++j) acc[i][j] = z;
  }

  const bf16_t* ap = og  + (mt * 64 + l15) * 256 + half * 16;
  const bf16_t* bp = wto + (nt * 64 + l15) * 256 + half * 16;

#pragma unroll
  for (int kk = 0; kk < 256; kk += 32) {
    bf16x16 a[4], b[4];
#pragma unroll
    for (int i = 0; i < 4; ++i) a[i] = load_afrag(ap + i * 16 * 256 + kk);
#pragma unroll
    for (int j = 0; j < 4; ++j) b[j] = load_bfrag(bp + j * 16 * 256 + kk);
#pragma unroll
    for (int i = 0; i < 4; ++i)
#pragma unroll
      for (int j = 0; j < 4; ++j) acc[i][j] = wmma_bf16(a[i], b[j], acc[i][j]);
  }

#pragma unroll
  for (int i = 0; i < 4; ++i) {
#pragma unroll
    for (int j = 0; j < 4; ++j) {
      const int n = nt * 64 + j * 16 + l15;
#pragma unroll
      for (int v = 0; v < 8; ++v) {
        int m = mt * 64 + i * 16 + v + 8 * half;
        out[m * 256 + n] = acc[i][j][v] + bo[n];
      }
    }
  }
}

extern "C" void kernel_launch(void* const* d_in, const int* in_sizes, int n_in,
                              void* d_out, int out_size, void* d_ws, size_t ws_size,
                              hipStream_t stream) {
  (void)in_sizes; (void)n_in; (void)out_size; (void)ws_size;
  const float* x    = (const float*)d_in[0];
  const float* mask = (const float*)d_in[1];
  const float* bias = (const float*)d_in[2];
  const float* Wq   = (const float*)d_in[3];
  const float* Wk   = (const float*)d_in[4];
  const float* Wv   = (const float*)d_in[5];
  const float* Wg   = (const float*)d_in[6];
  const float* bg   = (const float*)d_in[7];
  const float* Wo   = (const float*)d_in[8];
  const float* bo   = (const float*)d_in[9];
  float* out = (float*)d_out;

  // workspace bump allocator (256B aligned)
  char* p = (char*)d_ws;
  auto balloc = [&](size_t bytes) -> char* {
    char* r = p;
    p += (bytes + 255) & ~(size_t)255;
    return r;
  };
  bf16_t* xb   = (bf16_t*)balloc((size_t)NROWS * 256 * 2);  // 16.8 MB
  bf16_t* wtq  = (bf16_t*)balloc(256 * 256 * 2);
  bf16_t* wtk  = (bf16_t*)balloc(256 * 256 * 2);
  bf16_t* wtv  = (bf16_t*)balloc(256 * 256 * 2);
  bf16_t* wtg  = (bf16_t*)balloc(256 * 256 * 2);
  bf16_t* wto  = (bf16_t*)balloc(256 * 256 * 2);
  bf16_t* qb   = (bf16_t*)balloc((size_t)NROWS * 256 * 2);  // [r][h][s][d]
  bf16_t* kb   = (bf16_t*)balloc((size_t)NROWS * 256 * 2);  // [r][h][s][d]
  bf16_t* vtb  = (bf16_t*)balloc((size_t)NROWS * 256 * 2);  // [r][h][d][s]
  float*  gatef= (float*) balloc((size_t)NROWS * 256 * 4);  // 33.5 MB
  bf16_t* ogb  = (bf16_t*)balloc((size_t)NROWS * 256 * 2);

  // 1) convert activations, transpose+convert weights (Q scale folded in)
  cvt_kernel<<<(NROWS * 256) / 256, 256, 0, stream>>>(x, xb, NROWS * 256);
  wtrans_kernel<<<256, 256, 0, stream>>>(Wq, wtq, 0.17677669529663687f); // 1/sqrt(32)
  wtrans_kernel<<<256, 256, 0, stream>>>(Wk, wtk, 1.0f);
  wtrans_kernel<<<256, 256, 0, stream>>>(Wv, wtv, 1.0f);
  wtrans_kernel<<<256, 256, 0, stream>>>(Wg, wtg, 1.0f);
  wtrans_kernel<<<256, 256, 0, stream>>>(Wo, wto, 1.0f);

  // 2) projections: grid = (64-row token tiles, 64-col tiles, {Q,K,V,gate})
  proj_kernel<<<dim3(NROWS / 64, DMODEL / 64, 4), 32, 0, stream>>>(
      xb, wtq, wtk, wtv, wtg, bg, qb, kb, vtb, gatef);

  // 3) flash attention: 4 waves/block, 64 q-rows per block
  attn_kernel<<<dim3(SDIM / 64, NUM_HEADS, RDIM), 128, 0, stream>>>(
      qb, kb, vtb, bias, mask, gatef, ogb);

  // 4) output projection
  outgemm_kernel<<<dim3(NROWS / 64, DMODEL / 64), 32, 0, stream>>>(ogb, wto, bo, out);
}
